// SparseAttention_27882927685720
// MI455X (gfx1250) — matrix operand order repaired and checked by reading, not compiled, not run on hardware
//
#include <hip/hip_runtime.h>
#include <hip/hip_bf16.h>
#include <stdint.h>

#define B_   2
#define L_   2048
#define D_   2048
#define H_   16
#define DH_  128
#define WIN_ 512

typedef __bf16 bf16_t;
typedef bf16_t v16bf __attribute__((ext_vector_type(16)));
typedef float  v8f   __attribute__((ext_vector_type(8)));
typedef unsigned v4u __attribute__((ext_vector_type(4)));
typedef int      v8i __attribute__((ext_vector_type(8)));
typedef int      v4i __attribute__((ext_vector_type(4)));

union Frag {
  v16bf v;
  uint4 q[2];
};

__device__ __forceinline__ unsigned short f2bf(float f) {
  union { float f; unsigned u; } cv; cv.f = f;
  unsigned u = cv.u;
  unsigned r = u + 0x7FFFu + ((u >> 16) & 1u);   // round-to-nearest-even
  return (unsigned short)(r >> 16);
}
__device__ __forceinline__ float bf2f(unsigned short h) {
  union { unsigned u; float f; } cv; cv.u = ((unsigned)h) << 16; return cv.f;
}

__device__ __forceinline__ v8f zero8() {
  v8f z = {0.f, 0.f, 0.f, 0.f, 0.f, 0.f, 0.f, 0.f};
  return z;
}

// A-frag: 16x32 bf16, lane m(0..15)/m+16 holds row m, K halves {0-7,16-23}/{8-15,24-31}
__device__ __forceinline__ Frag loadA(const unsigned short* base, int ld, int row,
                                      int k0, int lane) {
  Frag f;
  const unsigned short* p = base + (size_t)row * ld + k0 + ((lane >> 4) << 3);
  f.q[0] = *(const uint4*)(p);
  f.q[1] = *(const uint4*)(p + 16);
  return f;
}
// B-frag: 32x16 bf16 K-major, lane n(0..15)/n+16 holds column n, K halves {0-7,16-23}/{8-15,24-31}
__device__ __forceinline__ Frag loadB(const unsigned short* base, int ld, int k0,
                                      int n0, int lane) {
  Frag f;
  unsigned short t[16];
  const unsigned short* p = base + (size_t)(k0 + ((lane >> 4) << 3)) * ld + n0 + (lane & 15);
#pragma unroll
  for (int s = 0; s < 16; ++s) t[s] = p[(size_t)((s & 7) + ((s >> 3) << 4)) * ld];
  __builtin_memcpy(&f, t, 32);
  return f;
}
// B-frag from LDS-staged K-major tile (row stride in elements), same K order as loadB
__device__ __forceinline__ Frag loadBlds(const unsigned short* base, int ld,
                                         int n0, int lane) {
  Frag f;
  unsigned short t[16];
  const unsigned short* p = base + (size_t)((lane >> 4) << 3) * ld + n0 + (lane & 15);
#pragma unroll
  for (int s = 0; s < 16; ++s) t[s] = p[(size_t)((s & 7) + ((s >> 3) << 4)) * ld];
  __builtin_memcpy(&f, t, 32);
  return f;
}

__device__ __forceinline__ v8f wmma_bf16(Frag a, Frag b, v8f c) {
  return __builtin_amdgcn_wmma_f32_16x16x32_bf16(false, a.v, false, b.v,
                                                 (short)0, c, false, false);
}

#define HAVE_TDM __has_builtin(__builtin_amdgcn_tensor_load_to_lds)

#if HAVE_TDM
// Issue one TDM transfer: 32(k) x 128(n) bf16 tile of Wt[2048x2048] -> LDS.
// pad_enable: +4 DWORDs after every 64 DWORDs -> LDS row stride 272B (136 elems),
// which spreads the lane-per-N 16-bit B-frag reads across banks.
__device__ __forceinline__ void tdm_issue_tile(const unsigned short* gsrc,
                                               unsigned ldsOff) {
  unsigned long long ga = (unsigned long long)(uintptr_t)gsrc;
  v4u g0;
  g0.x = 1u;                                        // count=1 (valid user D#)
  g0.y = ldsOff;                                    // lds_addr (bytes)
  g0.z = (unsigned)(ga & 0xFFFFFFFFu);              // global_addr[31:0]
  g0.w = (unsigned)((ga >> 32) & 0x01FFFFFFu)       // global_addr[56:32]
         | 0x80000000u;                             // type=2 ("image")
  v8i g1;
  g1[0] = (int)((1u << 16)      // data_size = 1 -> 2 bytes
                | (1u << 20)    // pad_enable
                | (5u << 22)    // pad_interval: 64 DWORDs (=256B row)
                | (3u << 25));  // pad_amount: 4 DWORDs (=16B)
  g1[1] = (int)(2048u << 16);   // tensor_dim0[15:0]=2048 in bits[63:48]
  g1[2] = (int)(2048u << 16);   // tensor_dim1[15:0]=2048 in bits[95:80]
  g1[3] = (int)(128u << 16);    // tile_dim0=128 in bits[127:112]
  g1[4] = 32;                   // tile_dim1=32, tile_dim2=0 (2D tile)
  g1[5] = 2048;                 // tensor_dim0_stride[31:0] = 2048 elements
  g1[6] = 0;
  g1[7] = 0;
  v4i gz4 = {0, 0, 0, 0};
  v8i gz8 = {0, 0, 0, 0, 0, 0, 0, 0};
  // clang-23 / therock-10.0 form: (g0, g1, g2, g3, g4, cpol)
  __builtin_amdgcn_tensor_load_to_lds(g0, g1, gz4, gz4, gz8, 0);
}
#endif

// ---------------------------------------------------------------------------
// dst[c][r] = (c < cols) ? src[r][c] : 0   ; dst is [colsPad x rows] bf16
// ---------------------------------------------------------------------------
__global__ void transpose_cast_kernel(const float* __restrict__ src,
                                      unsigned short* __restrict__ dst,
                                      int rows, int cols, int colsPad) {
  int idx = blockIdx.x * blockDim.x + threadIdx.x;
  int total = colsPad * rows;
  if (idx >= total) return;
  int c = idx / rows;
  int r = idx - c * rows;
  float v = (c < cols) ? src[(size_t)r * cols + c] : 0.0f;
  dst[(size_t)c * rows + r] = f2bf(v);
}

// ---------------------------------------------------------------------------
// RMSNorm: one 256-thread block per row of D=2048, output bf16
// ---------------------------------------------------------------------------
__global__ void rmsnorm_kernel(const float* __restrict__ x,
                               const float* __restrict__ w,
                               unsigned short* __restrict__ h) {
  int row = blockIdx.x;
  const float* xr = x + (size_t)row * D_;
  float v[8];
  float ss = 0.f;
#pragma unroll
  for (int i = 0; i < 8; ++i) {
    v[i] = xr[threadIdx.x + i * 256];
    ss += v[i] * v[i];
  }
#pragma unroll
  for (int m = 16; m >= 1; m >>= 1) ss += __shfl_xor(ss, m, 32);
  __shared__ float red[8];
  int wave = threadIdx.x >> 5, lane = threadIdx.x & 31;
  if (lane == 0) red[wave] = ss;
  __syncthreads();
  float tot = 0.f;
#pragma unroll
  for (int i = 0; i < 8; ++i) tot += red[i];
  float inv = rsqrtf(tot * (1.0f / (float)D_) + 1e-6f);
  unsigned short* hr = h + (size_t)row * D_;
#pragma unroll
  for (int i = 0; i < 8; ++i)
    hr[threadIdx.x + i * 256] = f2bf(v[i] * inv * w[threadIdx.x + i * 256]);
}

// ---------------------------------------------------------------------------
// LoRA stage 1: out[M x 32] = scale * (A[M x 2048] @ Bt[2048 x 16]), cols 16..31 = 0
// ---------------------------------------------------------------------------
__global__ void lora16_kernel(const unsigned short* __restrict__ A,
                              const unsigned short* __restrict__ Bt,
                              unsigned short* __restrict__ outp,
                              float scale) {
  int lane = threadIdx.x & 31;
  int wave = threadIdx.x >> 5;
  int m0 = (blockIdx.x * 4 + wave) * 16;
  v8f acc = zero8();
  int rowA = m0 + (lane & 15);
  for (int k0 = 0; k0 < D_; k0 += 32) {
    Frag a = loadA(A, D_, rowA, k0, lane);
    Frag b = loadB(Bt, 16, k0, 0, lane);
    acc = wmma_bf16(a, b, acc);
  }
  int half = lane >> 4, col = lane & 15;
#pragma unroll
  for (int g = 0; g < 8; ++g) {
    int row = m0 + g + 8 * half;
    outp[(size_t)row * 32 + col] = f2bf(acc[g] * scale);
    outp[(size_t)row * 32 + 16 + col] = 0;   // zero pad (rank 16 -> K=32)
  }
}

// ---------------------------------------------------------------------------
// Main GEMM: C[M=4096 x N=2048] = A @ Wt + hA @ Bt  (LoRA fused as one k-step).
// Weight k-slabs (32x128 bf16) are staged into LDS by the Tensor Data Mover
// (double buffered, TENSORcnt-synchronized); A-frags stream from global/L2.
// 8 waves (2 M x 4 N), wave computes 32x32, WG tile 64x128.
// ---------------------------------------------------------------------------
__global__ void proj_gemm_kernel(const unsigned short* __restrict__ A,
                                 const unsigned short* __restrict__ Wt,
                                 const unsigned short* __restrict__ hA,
                                 const unsigned short* __restrict__ Bt,
                                 const float* __restrict__ resid,
                                 float* __restrict__ outF,
                                 unsigned short* __restrict__ outB,
                                 int storeF32) {
  __shared__ __align__(16) unsigned short Wlds[2][32 * 136];  // 272B row stride
  int lane = threadIdx.x & 31;
  int wave = threadIdx.x >> 5;
  int wm = wave & 1, wn = wave >> 1;
  int m0 = blockIdx.y * 64 + wm * 32;
  int n0 = blockIdx.x * 128;
  int nw = wn * 32;
  v8f acc[2][2];
  acc[0][0] = zero8(); acc[0][1] = zero8();
  acc[1][0] = zero8(); acc[1][1] = zero8();
  int r0 = m0 + (lane & 15);
  const int NK = D_ / 32;

#if HAVE_TDM
  if (wave == 0) {
    tdm_issue_tile(Wt + (size_t)0 * D_ + n0,  (unsigned)(uintptr_t)&Wlds[0][0]);
    tdm_issue_tile(Wt + (size_t)32 * D_ + n0, (unsigned)(uintptr_t)&Wlds[1][0]);
  }
  for (int ki = 0; ki < NK; ++ki) {
    int k0 = ki * 32;
    if (wave == 0) {
      if (ki < NK - 1) __builtin_amdgcn_s_wait_tensorcnt(1);
      else             __builtin_amdgcn_s_wait_tensorcnt(0);
    }
    __syncthreads();
    if (k0 + 32 < D_)   // prefetch next A k-slab (global_prefetch_b8)
      __builtin_prefetch(A + (size_t)r0 * D_ + k0 + 32, 0, 0);
    Frag a0 = loadA(A, D_, r0, k0, lane);
    Frag a1 = loadA(A, D_, r0 + 16, k0, lane);
    const unsigned short* Wb = Wlds[ki & 1];
    Frag b0 = loadBlds(Wb, 136, nw, lane);
    Frag b1 = loadBlds(Wb, 136, nw + 16, lane);
    acc[0][0] = wmma_bf16(a0, b0, acc[0][0]);
    acc[0][1] = wmma_bf16(a0, b1, acc[0][1]);
    acc[1][0] = wmma_bf16(a1, b0, acc[1][0]);
    acc[1][1] = wmma_bf16(a1, b1, acc[1][1]);
    __syncthreads();
    if (wave == 0 && ki + 2 < NK)
      tdm_issue_tile(Wt + (size_t)(k0 + 64) * D_ + n0,
                     (unsigned)(uintptr_t)&Wlds[ki & 1][0]);
  }
#else
  // Fallback: cooperative synchronous staging (256 thr stage 32x128 per step)
  for (int ki = 0; ki < NK; ++ki) {
    int k0 = ki * 32;
    __syncthreads();
    {
      int srow = threadIdx.x >> 3;
      int scol = (threadIdx.x & 7) * 16;
      const uint4* gp = (const uint4*)(Wt + (size_t)(k0 + srow) * D_ + n0 + scol);
      uint4* lp = (uint4*)(&Wlds[0][srow * 136 + scol]);
      lp[0] = gp[0];
      lp[1] = gp[1];
    }
    __syncthreads();
    Frag a0 = loadA(A, D_, r0, k0, lane);
    Frag a1 = loadA(A, D_, r0 + 16, k0, lane);
    Frag b0 = loadBlds(Wlds[0], 136, nw, lane);
    Frag b1 = loadBlds(Wlds[0], 136, nw + 16, lane);
    acc[0][0] = wmma_bf16(a0, b0, acc[0][0]);
    acc[0][1] = wmma_bf16(a0, b1, acc[0][1]);
    acc[1][0] = wmma_bf16(a1, b0, acc[1][0]);
    acc[1][1] = wmma_bf16(a1, b1, acc[1][1]);
  }
#endif
  { // LoRA rank-16 (padded to K=32) epilogue
    Frag a0 = loadA(hA, 32, r0, 0, lane);
    Frag a1 = loadA(hA, 32, r0 + 16, 0, lane);
    Frag b0 = loadB(Bt, D_, 0, n0 + nw, lane);
    Frag b1 = loadB(Bt, D_, 0, n0 + nw + 16, lane);
    acc[0][0] = wmma_bf16(a0, b0, acc[0][0]);
    acc[0][1] = wmma_bf16(a0, b1, acc[0][1]);
    acc[1][0] = wmma_bf16(a1, b0, acc[1][0]);
    acc[1][1] = wmma_bf16(a1, b1, acc[1][1]);
  }
  int half = lane >> 4, col = lane & 15;
#pragma unroll
  for (int i = 0; i < 2; ++i)
#pragma unroll
    for (int j = 0; j < 2; ++j)
#pragma unroll
      for (int g = 0; g < 8; ++g) {
        int row = m0 + i * 16 + g + 8 * half;
        int cc = n0 + nw + j * 16 + col;
        float f = acc[i][j][g];
        if (storeF32)
          outF[(size_t)row * D_ + cc] = f + resid[(size_t)row * D_ + cc];
        else
          outB[(size_t)row * D_ + cc] = f2bf(f);
      }
}

// ---------------------------------------------------------------------------
// RoPE: z=0 -> Q in place ([B*L, H*DH]); z=1 -> K -> Kt[(b*H+h)*DH + d][l] (K-major)
// ---------------------------------------------------------------------------
__global__ void rope_kernel(unsigned short* __restrict__ Qb,
                            const unsigned short* __restrict__ Kbl,
                            unsigned short* __restrict__ Kt) {
  int idx = blockIdx.x * blockDim.x + threadIdx.x;   // B*L*H*64 = 4194304
  int i = idx & 63;
  int t = idx >> 6;
  int hh = t & (H_ - 1);
  t >>= 4;
  int l = t & (L_ - 1);
  int b = t >> 11;
  float theta = __powf(10000.0f, -((float)i) * (1.0f / 64.0f));
  float ang = (float)l * theta;
  float sn, cs;
  __sincosf(ang, &sn, &cs);
  size_t base = ((size_t)(b * L_ + l)) * D_ + hh * DH_;
  if (blockIdx.z == 0) {
    float x1 = bf2f(Qb[base + i]);
    float x2 = bf2f(Qb[base + 64 + i]);
    Qb[base + i]      = f2bf(x1 * cs - x2 * sn);
    Qb[base + 64 + i] = f2bf(x2 * cs + x1 * sn);
  } else {
    float x1 = bf2f(Kbl[base + i]);
    float x2 = bf2f(Kbl[base + 64 + i]);
    Kt[((size_t)((b * H_ + hh) * DH_ + i)) * L_ + l]      = f2bf(x1 * cs - x2 * sn);
    Kt[((size_t)((b * H_ + hh) * DH_ + i + 64)) * L_ + l] = f2bf(x2 * cs + x1 * sn);
  }
}

// ---------------------------------------------------------------------------
// Sliding-window causal flash attention. 8 waves/block; each wave owns a
// 16-query tile. S = Q.K^T via 4 WMMAs; online softmax (shfl within 16-lane
// halves); P staged through LDS (C-layout -> A-layout) then P.V via 8 WMMAs.
// ---------------------------------------------------------------------------
__global__ void attn_kernel(const unsigned short* __restrict__ Qb,
                            const unsigned short* __restrict__ Kt,
                            const unsigned short* __restrict__ Vb,
                            unsigned short* __restrict__ Outb) {
  __shared__ __align__(16) unsigned short Plds[8][16 * 40];  // padded stride 40
  int lane = threadIdx.x & 31;
  int wave = threadIdx.x >> 5;
  int q0 = (blockIdx.x * 8 + wave) * 16;
  int bh = blockIdx.y;
  int b = bh >> 4;
  int hh = bh & (H_ - 1);
  int half = lane >> 4, col = lane & 15;
  const int rowBase = b * L_;
  const unsigned short* KtH = Kt + (size_t)bh * DH_ * L_;
  unsigned short* P = Plds[wave];

  Frag qf[4];
  {
    int qrow = rowBase + q0 + (lane & 15);
#pragma unroll
    for (int s = 0; s < 4; ++s)
      qf[s] = loadA(Qb, D_, qrow, hh * DH_ + s * 32, lane);
  }

  float m8[8], l8[8];
  v8f o[8];
#pragma unroll
  for (int g = 0; g < 8; ++g) { m8[g] = -1e30f; l8[g] = 0.f; }
#pragma unroll
  for (int nt = 0; nt < 8; ++nt) o[nt] = zero8();

  const float sc = 0.08838834764831845f;  // 1/sqrt(128)
  int qLast = q0 + 15;
  int kc0 = q0 - (WIN_ - 1);
  if (kc0 < 0) kc0 = 0;
  kc0 &= ~31;

  for (int kc = kc0; kc <= qLast; kc += 32) {
    for (int sub = 0; sub < 2; ++sub) {
      int k0 = kc + sub * 16;
      v8f s = zero8();
#pragma unroll
      for (int st = 0; st < 4; ++st) {
        Frag kf = loadB(KtH, L_, st * 32, k0, lane);
        s = wmma_bf16(qf[st], kf, s);
      }
      int colk = k0 + col;
#pragma unroll
      for (int g = 0; g < 8; ++g) {
        int row = q0 + g + 8 * half;
        bool valid = (colk <= row) && (colk + WIN_ > row);
        float v = valid ? s[g] * sc : -1e30f;
        float r = v;
        r = fmaxf(r, __shfl_xor(r, 1, 32));
        r = fmaxf(r, __shfl_xor(r, 2, 32));
        r = fmaxf(r, __shfl_xor(r, 4, 32));
        r = fmaxf(r, __shfl_xor(r, 8, 32));
        float newm = fmaxf(m8[g], r);
        float corr = __expf(m8[g] - newm);
        m8[g] = newm;
        float p = valid ? __expf(v - newm) : 0.0f;
        float rs = p;
        rs += __shfl_xor(rs, 1, 32);
        rs += __shfl_xor(rs, 2, 32);
        rs += __shfl_xor(rs, 4, 32);
        rs += __shfl_xor(rs, 8, 32);
        l8[g] = l8[g] * corr + rs;
#pragma unroll
        for (int nt = 0; nt < 8; ++nt) o[nt][g] *= corr;
        P[(g + 8 * half) * 40 + sub * 16 + col] = f2bf(p);
      }
    }
    asm volatile("s_wait_dscnt 0" ::: "memory");  // P stores visible before loads
    Frag pf = loadA(P, 40, (lane & 15), 0, lane);
#pragma unroll
    for (int nt = 0; nt < 8; ++nt) {
      Frag vf = loadB(Vb, D_, rowBase + kc, hh * DH_ + nt * 16, lane);
      o[nt] = wmma_bf16(pf, vf, o[nt]);
    }
  }

#pragma unroll
  for (int g = 0; g < 8; ++g) {
    int row = q0 + g + 8 * half;
    float invl = 1.0f / l8[g];
#pragma unroll
    for (int nt = 0; nt < 8; ++nt)
      Outb[(size_t)(rowBase + row) * D_ + hh * DH_ + nt * 16 + col] =
          f2bf(o[nt][g] * invl);
  }
}

// ---------------------------------------------------------------------------
extern "C" void kernel_launch(void* const* d_in, const int* in_sizes, int n_in,
                              void* d_out, int out_size, void* d_ws, size_t ws_size,
                              hipStream_t stream) {
  const float* x  = (const float*)d_in[0];
  const float* nw = (const float*)d_in[1];
  const float* wq = (const float*)d_in[2];
  const float* wk = (const float*)d_in[3];
  const float* wv = (const float*)d_in[4];
  const float* wo = (const float*)d_in[5];
  const float* qA = (const float*)d_in[6];
  const float* qB = (const float*)d_in[7];
  const float* kA = (const float*)d_in[8];
  const float* kB = (const float*)d_in[9];
  const float* vA = (const float*)d_in[10];
  const float* vB = (const float*)d_in[11];
  const float* oA = (const float*)d_in[12];
  const float* oB = (const float*)d_in[13];
  float* out = (float*)d_out;

  char* w = (char*)d_ws;
  auto US = [&](size_t off) { return (unsigned short*)(w + off); };
  unsigned short* h    = US(0);           // 16 MB  (reused as Kt after projections)
  unsigned short* wqT  = US(16777216);
  unsigned short* wkT  = US(25165824);
  unsigned short* wvT  = US(33554432);
  unsigned short* woT  = US(41943040);
  unsigned short* qAt  = US(50331648);
  unsigned short* kAt  = US(50397184);
  unsigned short* vAt  = US(50462720);
  unsigned short* oAt  = US(50528256);
  unsigned short* qBt  = US(50593792);
  unsigned short* kBt  = US(50724864);
  unsigned short* vBt  = US(50855936);
  unsigned short* oBt  = US(50987008);
  unsigned short* hAq  = US(51118080);
  unsigned short* hAk  = US(51380224);
  unsigned short* hAv  = US(51642368);
  unsigned short* oHA  = US(51904512);
  unsigned short* Qb   = US(52166656);
  unsigned short* Kbl  = US(68943872);    // pre-RoPE K; reused as attn_out
  unsigned short* Vb   = US(85721088);
  unsigned short* Kt   = h;               // alias: h dead after projections
  unsigned short* attn = Kbl;             // alias: Kbl dead after rope

  auto tc = [&](const float* s, unsigned short* d, int rows, int cols, int colsPad) {
    int total = colsPad * rows;
    transpose_cast_kernel<<<(total + 255) / 256, 256, 0, stream>>>(s, d, rows, cols, colsPad);
  };
  tc(wq, wqT, 2048, 2048, 2048);
  tc(wk, wkT, 2048, 2048, 2048);
  tc(wv, wvT, 2048, 2048, 2048);
  tc(wo, woT, 2048, 2048, 2048);
  tc(qA, qAt, 16, 2048, 2048);
  tc(kA, kAt, 16, 2048, 2048);
  tc(vA, vAt, 16, 2048, 2048);
  tc(oA, oAt, 16, 2048, 2048);
  tc(qB, qBt, 2048, 16, 32);
  tc(kB, kBt, 2048, 16, 32);
  tc(vB, vBt, 2048, 16, 32);
  tc(oB, oBt, 2048, 16, 32);

  rmsnorm_kernel<<<4096, 256, 0, stream>>>(x, nw, h);

  const float ls = 1.0f / 16.0f;
  lora16_kernel<<<64, 128, 0, stream>>>(h, qAt, hAq, ls);
  lora16_kernel<<<64, 128, 0, stream>>>(h, kAt, hAk, ls);
  lora16_kernel<<<64, 128, 0, stream>>>(h, vAt, hAv, ls);

  dim3 pg(16, 64, 1);
  proj_gemm_kernel<<<pg, 256, 0, stream>>>(h, wqT, hAq, qBt, nullptr, nullptr, Qb, 0);
  proj_gemm_kernel<<<pg, 256, 0, stream>>>(h, wkT, hAk, kBt, nullptr, nullptr, Kbl, 0);
  proj_gemm_kernel<<<pg, 256, 0, stream>>>(h, wvT, hAv, vBt, nullptr, nullptr, Vb, 0);

  rope_kernel<<<dim3(16384, 1, 2), 256, 0, stream>>>(Qb, Kbl, Kt);

  attn_kernel<<<dim3(16, 32), 256, 0, stream>>>(Qb, Kt, Vb, attn);

  lora16_kernel<<<64, 128, 0, stream>>>(attn, oAt, oHA, ls);
  proj_gemm_kernel<<<pg, 256, 0, stream>>>(attn, woT, oHA, oBt, x, out, nullptr, 1);
}
